// LinearProbe_84799834292556
// MI455X (gfx1250) — matrix operand order repaired and checked by reading, not compiled, not run on hardware
//
#include <hip/hip_runtime.h>

// MI455X / gfx1250 — wave32, bf16 WMMA + async global->LDS (ASYNCcnt) pipeline.
// GEMM [T*B, K] x [K, C] fused with integrate-and-fire scan over T and mean.

typedef __attribute__((ext_vector_type(16))) __bf16 v16bf;
typedef __attribute__((ext_vector_type(4)))  __bf16 v4bf;
typedef __attribute__((ext_vector_type(8)))  float  v8f;
typedef __attribute__((ext_vector_type(4)))  float  v4f;

#define T_STEPS 8
#define BATCH   2048
#define INF     2048
#define NUMC    1000
#define KT      64            // bf16 K per outer iteration (2 x 16x16x32 WMMA)
#define COLS_WG 64            // 4 waves x 16 cols
#define THREADS 128
#define NITER   (INF / KT)    // 32

// ---------- helpers ----------

// float4 -> packed bf16x4 (RNE) -> 8 bytes (packed cvt HW).
static __device__ __forceinline__ uint2 cvt4bf(v4f f) {
    union { v4bf b; uint2 u; } x;
    x.b = __builtin_convertvector(f, v4bf);
    return x.u;
}

// Async DMA 16 bytes global -> LDS per lane (tracked by ASYNCcnt).
// VDST = 32-bit LDS byte offset (low 32 bits of shared-aperture flat address,
// per ISA 10.2 aperture mapping), VADDR = 64-bit global address, GV mode.
static __device__ __forceinline__ void async_b128(unsigned lds_off, const void* g) {
    asm volatile("global_load_async_to_lds_b128 %0, %1, off"
                 :: "v"(lds_off), "v"(g) : "memory");
}

// ---------- kernel 1: f32 -> bf16 pre-conversion into workspace ----------

__global__ __launch_bounds__(256)
void cvt_bf16_kernel(const float* __restrict__ src, unsigned short* __restrict__ dst) {
    int i = blockIdx.x * 256 + threadIdx.x;            // one float4 per thread
    v4f f = reinterpret_cast<const v4f*>(src)[i];
    reinterpret_cast<uint2*>(dst)[i] = cvt4bf(f);
}

// ---------- kernel 2: async-pipelined WMMA GEMM + IF scan ----------

__global__ __launch_bounds__(THREADS)
void snn_gemm_async_kernel(const unsigned short* __restrict__ Fbf, // [T][B][INF] bf16
                           const unsigned short* __restrict__ Wbf, // [C][INF]    bf16
                           const float* __restrict__ bias,         // [C]
                           float* __restrict__ out)                // [B][C]
{
    __shared__ alignas(16) unsigned short ldsA[2][T_STEPS * 16 * KT]; // 2 x 16 KB
    __shared__ alignas(16) unsigned short ldsB[2][COLS_WG * KT];      // 2 x  8 KB

    const int tid    = threadIdx.x;
    const int wave   = tid >> 5;
    const int lane   = tid & 31;
    const int lhalf  = lane >> 4;
    const int lmod   = lane & 15;
    const int b_base = blockIdx.x * 16;
    const int c_base = blockIdx.y * COLS_WG;

    v8f acc[T_STEPS];
    const v8f vzero = {};
#pragma unroll
    for (int t = 0; t < T_STEPS; ++t) acc[t] = vzero;

    // Staging coordinates: each thread moves 16B chunks.
    const int aR = tid >> 3;   // 0..15 : row within 16-row tile / col group base
    const int aK = tid & 7;    // 0..7  : 16B (8 bf16) chunk within 64-K row
    const unsigned short* pA[T_STEPS];
#pragma unroll
    for (int t = 0; t < T_STEPS; ++t)
        pA[t] = Fbf + (size_t)t * BATCH * INF + (size_t)(b_base + aR) * INF + aK * 8;
    const unsigned short* pW[4];
#pragma unroll
    for (int j = 0; j < 4; ++j) {
        int cg = c_base + aR + j * 16;            // covers local cols 0..63
        cg = cg < NUMC ? cg : (NUMC - 1);         // clamp: never stored for col>=NUMC
        pW[j] = Wbf + (size_t)cg * INF + aK * 8;
    }

    // Issue one tile's 12 async B128 transfers (8 for A, 4 for W) per thread.
    auto issue_tile = [&](int k0, int buf) {
#pragma unroll
        for (int t = 0; t < T_STEPS; ++t) {
            unsigned off = (unsigned)(uintptr_t)&ldsA[buf][(t * 16 + aR) * KT + aK * 8];
            async_b128(off, pA[t] + k0);
        }
#pragma unroll
        for (int j = 0; j < 4; ++j) {
            unsigned off = (unsigned)(uintptr_t)&ldsB[buf][(aR + j * 16) * KT + aK * 8];
            async_b128(off, pW[j] + k0);
        }
    };

    issue_tile(0, 0);   // prologue: tile 0 in flight

    for (int it = 0; it < NITER; ++it) {
        const int cur = it & 1;
        __syncthreads();                          // all waves done reading buf cur^1
        if (it + 1 < NITER) {
            issue_tile((it + 1) * KT, cur ^ 1);   // prefetch next tile (12 per thread)
            asm volatile("s_wait_asynccnt 0xc" ::: "memory"); // tile `cur` landed
        } else {
            asm volatile("s_wait_asynccnt 0x0" ::: "memory");
        }
        __syncthreads();                          // cross-wave: buf cur visible

        const unsigned short* lA = ldsA[cur];
        const unsigned short* lB = ldsB[cur];
#pragma unroll
        for (int s = 0; s < 2; ++s) {
            // B fragment: two 16-byte chunks per lane (k = 8h.., 16+8h..)
            const int bOff = (wave * 16 + lmod) * KT + s * 32 + lhalf * 8;
            union { uint4 u[2]; v16bf v; } bc;
            bc.u[0] = *reinterpret_cast<const uint4*>(&lB[bOff]);
            bc.u[1] = *reinterpret_cast<const uint4*>(&lB[bOff + 16]);

            union { uint4 u[2]; v16bf v; } af[T_STEPS];
#pragma unroll
            for (int t = 0; t < T_STEPS; ++t) {
                int aBase = (t * 16 + lmod) * KT + s * 32 + lhalf * 8;
                af[t].u[0] = *reinterpret_cast<const uint4*>(&lA[aBase]);
                af[t].u[1] = *reinterpret_cast<const uint4*>(&lA[aBase + 16]);
            }
#pragma unroll
            for (int t = 0; t < T_STEPS; ++t) {
                acc[t] = __builtin_amdgcn_wmma_f32_16x16x32_bf16(
                    false, af[t].v, false, bc.v, (short)0, acc[t], false, false);
            }
        }
    }

    // ---- epilogue: bias + integrate-and-fire scan over T, write mean spikes ----
    const int col = c_base + wave * 16 + lmod;
    const float bv = (col < NUMC) ? bias[col] : 0.0f;
#pragma unroll
    for (int j = 0; j < 8; ++j) {
        int m = j + lhalf * 8;
        float v = 0.0f, ssum = 0.0f;
#pragma unroll
        for (int t = 0; t < T_STEPS; ++t) {
            v += acc[t][j] + bv;                    // charge
            float s = (v >= 1.0f) ? 1.0f : 0.0f;    // fire (V_TH = 1.0)
            ssum += s;
            v = (1.0f - s) * v;                     // hard reset
        }
        if (col < NUMC)
            out[(size_t)(b_base + m) * NUMC + col] = ssum * (1.0f / (float)T_STEPS);
    }
}

// ---------- fallback: fused in-loop conversion (no workspace needed) ----------

__global__ __launch_bounds__(THREADS)
void snn_linear_if_fallback(const float* __restrict__ F,
                            const float* __restrict__ W,
                            const float* __restrict__ bias,
                            float* __restrict__ out)
{
    __shared__ alignas(16) unsigned short ldsA[T_STEPS * 16 * KT];
    __shared__ alignas(16) unsigned short ldsB[COLS_WG * KT];

    const int tid    = threadIdx.x;
    const int wave   = tid >> 5;
    const int lane   = tid & 31;
    const int lhalf  = lane >> 4;
    const int lmod   = lane & 15;
    const int b_base = blockIdx.x * 16;
    const int c_base = blockIdx.y * COLS_WG;

    v8f acc[T_STEPS];
    const v8f vzero = {};
#pragma unroll
    for (int t = 0; t < T_STEPS; ++t) acc[t] = vzero;

    const int sR  = tid >> 4;
    const int sKq = tid & 15;
    const float* pA = F + (size_t)(b_base + sR) * INF + sKq * 4;
    const float* pW[8];
#pragma unroll
    for (int i = 0; i < 8; ++i) {
        int cg = c_base + sR + i * 8;
        cg = cg < NUMC ? cg : (NUMC - 1);
        pW[i] = W + (size_t)cg * INF + sKq * 4;
    }

    for (int k0 = 0; k0 < INF; k0 += KT) {
        __syncthreads();
#pragma unroll
        for (int t = 0; t < T_STEPS; ++t) {
#pragma unroll
            for (int i = 0; i < 2; ++i) {
                int r  = sR + i * 8;
                v4f f4 = *reinterpret_cast<const v4f*>(
                    pA + (size_t)t * BATCH * INF + (size_t)i * 8 * INF + k0);
                *reinterpret_cast<uint2*>(&ldsA[(t * 16 + r) * KT + sKq * 4]) = cvt4bf(f4);
            }
        }
#pragma unroll
        for (int i = 0; i < 8; ++i) {
            int c  = sR + i * 8;
            v4f f4 = *reinterpret_cast<const v4f*>(pW[i] + k0);
            *reinterpret_cast<uint2*>(&ldsB[(c * KT + sKq * 4)]) = cvt4bf(f4);
        }
        __syncthreads();
#pragma unroll
        for (int s = 0; s < 2; ++s) {
            const int bOff = (wave * 16 + lmod) * KT + s * 32 + lhalf * 8;
            union { uint4 u[2]; v16bf v; } bc;
            bc.u[0] = *reinterpret_cast<const uint4*>(&ldsB[bOff]);
            bc.u[1] = *reinterpret_cast<const uint4*>(&ldsB[bOff + 16]);
            union { uint4 u[2]; v16bf v; } af[T_STEPS];
#pragma unroll
            for (int t = 0; t < T_STEPS; ++t) {
                int aBase = (t * 16 + lmod) * KT + s * 32 + lhalf * 8;
                af[t].u[0] = *reinterpret_cast<const uint4*>(&ldsA[aBase]);
                af[t].u[1] = *reinterpret_cast<const uint4*>(&ldsA[aBase + 16]);
            }
#pragma unroll
            for (int t = 0; t < T_STEPS; ++t) {
                acc[t] = __builtin_amdgcn_wmma_f32_16x16x32_bf16(
                    false, af[t].v, false, bc.v, (short)0, acc[t], false, false);
            }
        }
    }

    const int col = c_base + wave * 16 + lmod;
    const float bv = (col < NUMC) ? bias[col] : 0.0f;
#pragma unroll
    for (int j = 0; j < 8; ++j) {
        int m = j + lhalf * 8;
        float v = 0.0f, ssum = 0.0f;
#pragma unroll
        for (int t = 0; t < T_STEPS; ++t) {
            v += acc[t][j] + bv;
            float s = (v >= 1.0f) ? 1.0f : 0.0f;
            ssum += s;
            v = (1.0f - s) * v;
        }
        if (col < NUMC)
            out[(size_t)(b_base + m) * NUMC + col] = ssum * (1.0f / (float)T_STEPS);
    }
}

// ---------- launcher ----------

extern "C" void kernel_launch(void* const* d_in, const int* in_sizes, int n_in,
                              void* d_out, int out_size, void* d_ws, size_t ws_size,
                              hipStream_t stream) {
    (void)in_sizes; (void)n_in; (void)out_size;
    const float* F    = (const float*)d_in[0];   // [T, B, INF]
    const float* W    = (const float*)d_in[1];   // [C, INF]
    const float* bias = (const float*)d_in[2];   // [C]
    float* out        = (float*)d_out;           // [B, C]

    const size_t fElems = (size_t)T_STEPS * BATCH * INF;   // 33,554,432
    const size_t wElems = (size_t)NUMC * INF;              //  2,048,000
    const size_t need   = (fElems + wElems) * sizeof(unsigned short);

    dim3 grid(BATCH / 16, (NUMC + COLS_WG - 1) / COLS_WG);  // 128 x 16
    dim3 block(THREADS);                                    // 4 waves (wave32)

    if (d_ws != nullptr && ws_size >= need) {
        unsigned short* Fbf = (unsigned short*)d_ws;
        unsigned short* Wbf = Fbf + fElems;
        cvt_bf16_kernel<<<(unsigned)(fElems / 4 / 256), 256, 0, stream>>>(F, Fbf);
        cvt_bf16_kernel<<<(unsigned)(wElems / 4 / 256), 256, 0, stream>>>(W, Wbf);
        snn_gemm_async_kernel<<<grid, block, 0, stream>>>(Fbf, Wbf, bias, out);
    } else {
        snn_linear_if_fallback<<<grid, block, 0, stream>>>(F, W, bias, out);
    }
}
